// HeteroGNN3_45543833207289
// MI455X (gfx1250) — matrix operand order repaired and checked by reading, not compiled, not run on hardware
//
#include <hip/hip_runtime.h>
#include <hip/hip_bf16.h>

typedef __attribute__((ext_vector_type(16))) _Float16 v16h;
typedef __attribute__((ext_vector_type(8)))  _Float16 v8h;
typedef __attribute__((ext_vector_type(8)))  float    v8f;

// ---------------- problem constants ----------------
#define N_ATOM 100000
#define N_AA   10000
#define E_AA2AT 40000
#define E_AT2AA 40000
#define E_AT2AT 80000

// ---------------- CDNA5 async copy (global -> LDS, 16B per lane) ----------
__device__ __forceinline__ void async_copy16(unsigned lds_off, const void* gptr) {
  unsigned long long ga = (unsigned long long)gptr;
  asm volatile("global_load_async_to_lds_b128 %0, %1, off" :: "v"(lds_off), "v"(ga) : "memory");
}
__device__ __forceinline__ void async_wait0() {
  asm volatile("s_wait_asynccnt 0" ::: "memory");
}
__device__ __forceinline__ unsigned lds_addr(const void* p) {
  return (unsigned)(size_t)p;  // low 32 bits of flat LDS address == LDS offset
}

// ---------------- WMMA fragment helpers -----------------------------------
// A 16x32 f16 fragment: per lane (m=lane&15, hi=lane>>4) the 16 halves are one
// contiguous 16-half run: A[m][kbase+hi*16 .. +15] (two 8-half loads).
__device__ __forceinline__ v16h load_a_frag(const _Float16* A, int lda, int kbase) {
  int lane = threadIdx.x & 31, m = lane & 15, hi = lane >> 4;
  const _Float16* p = A + m * lda + kbase + hi * 16;
  v8h lo = *(const v8h*)p;
  v8h hp = *(const v8h*)(p + 8);
  v16h a;
#pragma unroll
  for (int j = 0; j < 8; ++j) { a[j] = lo[j]; a[8 + j] = hp[j]; }
  return a;
}
// B 32x16 fragment from K-major (transposed) storage Bt[n*ldk + k]:
// lane (n=lane&15 + nbase, hi) needs Bt[n][kbase+hi*16 .. +15] -> one 32B run.
__device__ __forceinline__ v16h load_bT_frag(const _Float16* Bt, int ldk, int kbase, int nbase) {
  int lane = threadIdx.x & 31, n = (lane & 15) + nbase, hi = lane >> 4;
  return *(const v16h*)(Bt + n * ldk + kbase + hi * 16);
}

__device__ __forceinline__ unsigned fkeyU(float f) {
  unsigned u = __float_as_uint(f);
  return (u & 0x80000000u) ? ~u : (u | 0x80000000u);
}
__device__ __forceinline__ float fkeyInvU(unsigned k) {
  return (k & 0x80000000u) ? __uint_as_float(k & 0x7fffffffu) : __uint_as_float(~k);
}
__device__ __forceinline__ float selu_f(float x) {
  const float sc = 1.0507009873554805f, al = 1.6732632423543772f;
  return x > 0.f ? sc * x : sc * al * (__expf(x) - 1.f);
}

// ---------------- utility fills / weight pre-transforms ----------------
__global__ void fill_f32(float* p, float v, int n) {
  int i = blockIdx.x * 256 + threadIdx.x;
  if (i < n) p[i] = v;
}
__global__ void fill_u32(unsigned* p, unsigned v, int n) {
  int i = blockIdx.x * 256 + threadIdx.x;
  if (i < n) p[i] = v;
}
// out[n*Kpad + k] = (k<K) ? in[k*N + n] : 0   (f32 -> f16, K-major w/ padding)
__global__ void cvt_transpose_pad(const float* in, _Float16* out, int K, int Kpad, int N) {
  int i = blockIdx.x * 256 + threadIdx.x;
  if (i >= N * Kpad) return;
  int n = i / Kpad, k = i % Kpad;
  out[i] = (_Float16)((k < K) ? in[k * N + n] : 0.f);
}
// w3 [128 x 1024]: transpose each row's 32x32 block: out[c*1024 + o*32 + i] = in[c*1024 + i*32 + o]
__global__ void cvt_block_transpose32(const float* in, _Float16* out) {
  int i = blockIdx.x * 256 + threadIdx.x;
  if (i >= 128 * 1024) return;
  int c = i >> 10, r = i & 1023, ii = r >> 5, o = r & 31;
  out[c * 1024 + o * 32 + ii] = (_Float16)in[i];
}

// ---------------- edge MLP: layers 1+2 fused with LayerNorm (WMMA) ---------
// One block = 16 edges, 8 waves; weights pre-transposed f16, staged via async DMA.
template <int FE>
__global__ void mlp_l1l2(const float* __restrict__ ef, const _Float16* __restrict__ w1t,
                         const float* __restrict__ b1, const float* __restrict__ g,
                         const float* __restrict__ bt, const _Float16* __restrict__ w2t,
                         const float* __restrict__ b2, _Float16* __restrict__ h_out, int E) {
  __shared__ __align__(32) _Float16 sW1t[128 * 32];    // [n][k] 8KB
  __shared__ __align__(32) _Float16 sW2t[128 * 128];   // [n][k] 32KB
  __shared__ __align__(32) _Float16 sA[16 * 32];
  __shared__ float    sH[16 * 128];
  __shared__ __align__(32) _Float16 sH16[16 * 128];
  __shared__ float    rowMean[16], rowRstd[16];

  int tile = blockIdx.x, t = threadIdx.x, w = t >> 5, lane = t & 31;

  // async-stage weights (no VGPR round trip)
  {
    unsigned l1 = lds_addr(sW1t), l2 = lds_addr(sW2t);
#pragma unroll
    for (int i = t; i < 512; i += 256) async_copy16(l1 + i * 16, (const char*)w1t + i * 16);
#pragma unroll
    for (int i = t; i < 2048; i += 256) async_copy16(l2 + i * 16, (const char*)w2t + i * 16);
  }
  for (int i = t; i < 16 * 32; i += 256) {
    int m = i >> 5, c = i & 31, e = tile * 16 + m;
    sA[i] = (_Float16)((c < FE && e < E) ? ef[e * FE + c] : 0.f);
  }
  async_wait0();
  __syncthreads();

  // layer 1: [16,32] @ [32,128], K=32 -> 1 WMMA per wave
  {
    v8f acc = {};
    v16h a = load_a_frag(sA, 32, 0);
    v16h b = load_bT_frag(sW1t, 32, 0, 16 * w);
    acc = __builtin_amdgcn_wmma_f32_16x16x32_f16(false, a, false, b, (short)0, acc, false, false);
    int n = (lane & 15) + 16 * w, hi = lane >> 4;
#pragma unroll
    for (int r = 0; r < 8; ++r) sH[(r + hi * 8) * 128 + n] = fmaxf(acc[r] + b1[n], 0.f);
  }
  __syncthreads();
  if (t < 16) {  // LayerNorm stats per row
    float s = 0.f, q = 0.f;
    for (int c = 0; c < 128; ++c) { float v = sH[t * 128 + c]; s += v; q += v * v; }
    float mean = s / 128.f;
    rowMean[t] = mean;
    rowRstd[t] = rsqrtf(q / 128.f - mean * mean + 1e-5f);
  }
  __syncthreads();
  for (int i = t; i < 16 * 128; i += 256) {
    int m = i >> 7, c = i & 127;
    sH16[i] = (_Float16)((sH[i] - rowMean[m]) * rowRstd[m] * g[c] + bt[c]);
  }
  __syncthreads();

  // layer 2: [16,128] @ [128,128], K=128 -> 4 WMMA per wave
  {
    v8f acc = {};
#pragma unroll
    for (int kk = 0; kk < 4; ++kk) {
      v16h a = load_a_frag(sH16, 128, kk * 32);
      v16h b = load_bT_frag(sW2t, 128, kk * 32, 16 * w);
      acc = __builtin_amdgcn_wmma_f32_16x16x32_f16(false, a, false, b, (short)0, acc, false, false);
    }
    int n = (lane & 15) + 16 * w, hi = lane >> 4;
#pragma unroll
    for (int r = 0; r < 8; ++r) {
      int m = r + hi * 8, e = tile * 16 + m;
      if (e < E) h_out[e * 128 + n] = (_Float16)fmaxf(acc[r] + b2[n], 0.f);
    }
  }
}

// ---------------- fused layer-3 + NNConv per-edge matvec + scatter ---------
// msg_e = x_src[e] @ reshape(h_e @ w3 + b3, 32,32); A rows are rank-1 h(x)x
// built in registers; B = w3 row c, 32x32-block-transposed so each lane's
// fragment is one 32B contiguous global load (L2-resident, shared by edges).
__global__ void l3_nnconv(const _Float16* __restrict__ h, const float* __restrict__ x,
                          const int* __restrict__ src, const int* __restrict__ dst,
                          const _Float16* __restrict__ w3bt, const float* __restrict__ b3,
                          float* __restrict__ agg, float* __restrict__ cnt, int E) {
  __shared__ __align__(32) _Float16 sH[16 * 128];
  __shared__ __align__(32) _Float16 sX[16 * 32];
  __shared__ float sPart[8][16][32];

  int tile = blockIdx.x, t = threadIdx.x, w = t >> 5, lane = t & 31;
  {  // async-stage the h tile (f16, 16 rows x 256B); zero-fill tail rows
    unsigned lh = lds_addr(sH);
    int m = t >> 4, seg = t & 15, e = tile * 16 + m;
    if (e < E) async_copy16(lh + t * 16, (const char*)(h + e * 128) + seg * 16);
    else *(v8h*)(sH + m * 128 + seg * 8) = (v8h)(_Float16)0.f;
  }
  for (int i = t; i < 16 * 32; i += 256) {
    int m = i >> 5, k = i & 31, e = tile * 16 + m;
    sX[i] = (_Float16)((e < E) ? x[src[e] * 32 + k] : 0.f);
  }
  async_wait0();
  __syncthreads();

  v8f acc0 = {}, acc1 = {};
  int m = lane & 15, hi = lane >> 4, o = lane & 15;
  v8h xlo = *(const v8h*)(sX + m * 32 + hi * 16);        // loop-invariant
  v8h xhi = *(const v8h*)(sX + m * 32 + hi * 16 + 8);
  for (int cc = 0; cc < 16; ++cc) {
    int c = w * 16 + cc;
    const _Float16* bp = w3bt + c * 1024;
    if (cc + 1 < 16) __builtin_prefetch(w3bt + (c + 1) * 1024, 0, 1);
    _Float16 hv = sH[m * 128 + c];
    v16h a;
#pragma unroll
    for (int j = 0; j < 8; ++j) { a[j] = hv * xlo[j]; a[8 + j] = hv * xhi[j]; }
    v16h b0 = *(const v16h*)(bp + o * 32 + hi * 16);         // 2x b128
    v16h b1 = *(const v16h*)(bp + (o + 16) * 32 + hi * 16);  // 2x b128
    acc0 = __builtin_amdgcn_wmma_f32_16x16x32_f16(false, a, false, b0, (short)0, acc0, false, false);
    acc1 = __builtin_amdgcn_wmma_f32_16x16x32_f16(false, a, false, b1, (short)0, acc1, false, false);
  }
  {
#pragma unroll
    for (int r = 0; r < 8; ++r) {
      sPart[w][r + hi * 8][o]      = acc0[r];
      sPart[w][r + hi * 8][o + 16] = acc1[r];
    }
  }
  __syncthreads();
  for (int i = t; i < 512; i += 256) {
    int mm = i >> 5, oo = i & 31, e = tile * 16 + mm;
    if (e < E) {
      float s = 0.f;
#pragma unroll
      for (int ww = 0; ww < 8; ++ww) s += sPart[ww][mm][oo];
      float bias = 0.f;
      for (int ii = 0; ii < 32; ++ii) bias += (float)sX[mm * 32 + ii] * b3[ii * 32 + oo];
      int d = dst[e];
      atomicAdd(&agg[d * 32 + oo], s + bias);
      if (oo == 0) atomicAdd(&cnt[d], 1.0f);
    }
  }
}

// ---------------- segment-mean finalize + root + proj + ReLU + LN ----------
__global__ void finalize_proj(const float* agg1, const float* cnt1,
                              const float* agg2, const float* cnt2,
                              const float* __restrict__ x,
                              const float* root1, const float* bias1,
                              const float* root2, const float* bias2, float scale,
                              const float* pw, const float* pb, const float* pg,
                              const float* pbt, float* __restrict__ out, int N) {
  int n = blockIdx.x;
  if (n >= N) return;
  __shared__ float comb[32];
  __shared__ float rs[128], rq[128];
  int t = threadIdx.x;
  if (t < 32) {
    float v = agg1[n * 32 + t] / fmaxf(cnt1[n], 1.f);
    float r = 0.f;
    for (int k = 0; k < 32; ++k) r += x[n * 32 + k] * root1[k * 32 + t];
    v += r + bias1[t];
    if (agg2) {
      float v2 = agg2[n * 32 + t] / fmaxf(cnt2[n], 1.f);
      float r2 = 0.f;
      for (int k = 0; k < 32; ++k) r2 += x[n * 32 + k] * root2[k * 32 + t];
      v += v2 + r2 + bias2[t];
    }
    comb[t] = v * scale;
  }
  __syncthreads();
  float a = pb[t];
  for (int i = 0; i < 32; ++i) a += comb[i] * pw[i * 128 + t];
  float v = fmaxf(a, 0.f);
  rs[t] = v; rq[t] = v * v;
  __syncthreads();
  for (int s = 64; s > 0; s >>= 1) {
    if (t < s) { rs[t] += rs[t + s]; rq[t] += rq[t + s]; }
    __syncthreads();
  }
  float mean = rs[0] / 128.f;
  float var = rq[0] / 128.f - mean * mean;
  out[n * 128 + t] = (v - mean) * rsqrtf(var + 1e-5f) * pg[t] + pbt[t];
}

// ---------------- GAT projection xp = feat @ W  ([N,128]@[128,512], WMMA) --
// Bt = pre-transposed W (K-major); each block async-stages its 32KB panel.
__global__ void gemm_xp(const float* __restrict__ A, const _Float16* __restrict__ Bt,
                        _Float16* __restrict__ C, int M) {
  __shared__ __align__(32) _Float16 sBt[128 * 128];  // [n][k]
  __shared__ __align__(32) _Float16 sA[16 * 128];
  int tile = blockIdx.x, cg = blockIdx.y;  // 4 column groups (= heads)
  int t = threadIdx.x, w = t >> 5, lane = t & 31;
  {
    unsigned lb = lds_addr(sBt);
    const char* gp = (const char*)(Bt + cg * 128 * 128);
#pragma unroll
    for (int i = t; i < 2048; i += 256) async_copy16(lb + i * 16, gp + i * 16);
  }
  for (int i = t; i < 16 * 128; i += 256) {
    int m = i >> 7, k = i & 127, gm = tile * 16 + m;
    sA[i] = (_Float16)((gm < M) ? A[gm * 128 + k] : 0.f);
  }
  async_wait0();
  __syncthreads();
  v8f acc = {};
#pragma unroll
  for (int kk = 0; kk < 4; ++kk) {
    v16h a = load_a_frag(sA, 128, kk * 32);
    v16h b = load_bT_frag(sBt, 128, kk * 32, w * 16);
    acc = __builtin_amdgcn_wmma_f32_16x16x32_f16(false, a, false, b, (short)0, acc, false, false);
  }
  int n = lane & 15, hi = lane >> 4;
#pragma unroll
  for (int r = 0; r < 8; ++r) {
    int m = tile * 16 + r + hi * 8;
    if (m < M) C[m * 512 + cg * 128 + w * 16 + n] = (_Float16)acc[r];
  }
}

// ---------------- GAT attention logits ----------------
__global__ void gat_att(const _Float16* __restrict__ xp, const float* att_s,
                        const float* att_d, float* as, float* ad, int N) {
  int idx = blockIdx.x * 256 + threadIdx.x;
  if (idx >= N * 4) return;
  int n = idx >> 2, h = idx & 3;
  const _Float16* row = xp + (size_t)n * 512 + h * 128;
  float s1 = 0.f, s2 = 0.f;
  for (int c = 0; c < 128; ++c) {
    float v = (float)row[c];
    s1 += v * att_s[h * 128 + c];
    s2 += v * att_d[h * 128 + c];
  }
  as[idx] = s1; ad[idx] = s2;
}

// ---------------- GAT segment softmax (max / exp-sum / aggregate) ----------
__global__ void gat_edge_max(const int* src, const int* dst, const float* as,
                             const float* ad, unsigned* mx, int E, int N) {
  int idx = blockIdx.x * 256 + threadIdx.x;
  if (idx >= (E + N) * 4) return;
  int h = idx & 3, e = idx >> 2, s, d;
  if (e < E) { s = src[e]; d = dst[e]; } else { s = d = e - E; }  // self loops
  float v = as[s * 4 + h] + ad[d * 4 + h];
  v = v > 0.f ? v : 0.2f * v;  // leaky relu
  atomicMax(&mx[d * 4 + h], fkeyU(v));
}
__global__ void gat_edge_exp(const int* src, const int* dst, const float* as,
                             const float* ad, const unsigned* mx, float* ex,
                             float* den, int E, int N) {
  int idx = blockIdx.x * 256 + threadIdx.x;
  if (idx >= (E + N) * 4) return;
  int h = idx & 3, e = idx >> 2, s, d;
  if (e < E) { s = src[e]; d = dst[e]; } else { s = d = e - E; }
  float v = as[s * 4 + h] + ad[d * 4 + h];
  v = v > 0.f ? v : 0.2f * v;
  float xv = __expf(v - fkeyInvU(mx[d * 4 + h]));
  ex[idx] = xv;
  atomicAdd(&den[d * 4 + h], xv);
}
__global__ void gat_edge_aggr(const int* src, const int* dst, const float* ex,
                              const float* den, const _Float16* __restrict__ xp,
                              float* gout, int E, int N) {
  int idx = blockIdx.x * 256 + threadIdx.x;
  if (idx >= (E + N) * 4) return;
  int h = idx & 3, e = idx >> 2, s, d;
  if (e < E) { s = src[e]; d = dst[e]; } else { s = d = e - E; }
  float alpha = 0.25f * ex[idx] / den[d * 4 + h];  // fold head-mean (1/4)
  const _Float16* row = xp + (size_t)s * 512 + h * 128;
  float* o = gout + (size_t)d * 128;
  for (int c = 0; c < 128; ++c) atomicAdd(&o[c], alpha * (float)row[c]);
}

// ---------------- bias + global mean pool (one block per feature) ----------
__global__ void bias_pool(const float* __restrict__ gout, const float* b,
                          float* pooled, int N, int off) {
  int c = blockIdx.x;
  float s = 0.f;
  for (int n = threadIdx.x; n < N; n += 256) s += gout[(size_t)n * 128 + c];
  __shared__ float red[256];
  red[threadIdx.x] = s;
  __syncthreads();
  for (int st = 128; st > 0; st >>= 1) {
    if (threadIdx.x < st) red[threadIdx.x] += red[threadIdx.x + st];
    __syncthreads();
  }
  if (threadIdx.x == 0) pooled[off + c] = red[0] / (float)N + b[c];
}

// ---------------- classifier head (single block) ----------------
__global__ void clf_kernel(const float* pooled, const float* w1, const float* b1,
                           const float* w2, const float* b2, const float* w3,
                           const float* b3, float* out) {
  __shared__ float x[256], h1[256], h2[128];
  int t = threadIdx.x;
  x[t] = pooled[t];
  __syncthreads();
  float a = b1[t];
  for (int i = 0; i < 256; ++i) a += x[i] * w1[i * 256 + t];
  h1[t] = selu_f(a);
  __syncthreads();
  if (t < 128) {
    float a2 = b2[t];
    for (int i = 0; i < 256; ++i) a2 += h1[i] * w2[i * 128 + t];
    h2[t] = selu_f(a2);
  }
  __syncthreads();
  if (t == 0) {
    float a3 = b3[0];
    for (int i = 0; i < 128; ++i) a3 += h2[i] * w3[i];
    out[0] = a3;
  }
}

// ---------------- host launch ----------------
static inline int blks(long long n, int b) { return (int)((n + b - 1) / b); }

extern "C" void kernel_launch(void* const* d_in, const int* in_sizes, int n_in,
                              void* d_out, int out_size, void* d_ws, size_t ws_size,
                              hipStream_t stream) {
  // positional inputs
  const float* x_aa   = (const float*)d_in[0];
  const float* x_atom = (const float*)d_in[1];
  const int* ei1 = (const int*)d_in[2];
  const int* ei2 = (const int*)d_in[3];
  const int* ei3 = (const int*)d_in[4];
  const float* ea1 = (const float*)d_in[5];
  const float* ea2 = (const float*)d_in[6];
  const float* ea3 = (const float*)d_in[7];
  // params (pytree-sorted dict keys)
  const float* bias1 = (const float*)d_in[8];
  const float* bias2 = (const float*)d_in[9];
  const float* bias3 = (const float*)d_in[10];
  const float* c_b1 = (const float*)d_in[11];
  const float* c_b2 = (const float*)d_in[12];
  const float* c_b3 = (const float*)d_in[13];
  const float* c_w1 = (const float*)d_in[14];
  const float* c_w2 = (const float*)d_in[15];
  const float* c_w3 = (const float*)d_in[16];
  const float* gaa_ad = (const float*)d_in[17];
  const float* gaa_as = (const float*)d_in[18];
  const float* gaa_b  = (const float*)d_in[19];
  const float* gaa_w  = (const float*)d_in[20];
  const float* gat_ad = (const float*)d_in[21];
  const float* gat_as = (const float*)d_in[22];
  const float* gat_b  = (const float*)d_in[23];
  const float* gat_w  = (const float*)d_in[24];
  const float* m1_b1 = (const float*)d_in[25]; const float* m1_b2 = (const float*)d_in[26];
  const float* m1_b3 = (const float*)d_in[27]; const float* m1_bt = (const float*)d_in[28];
  const float* m1_g  = (const float*)d_in[29]; const float* m1_w1 = (const float*)d_in[30];
  const float* m1_w2 = (const float*)d_in[31]; const float* m1_w3 = (const float*)d_in[32];
  const float* m2_b1 = (const float*)d_in[33]; const float* m2_b2 = (const float*)d_in[34];
  const float* m2_b3 = (const float*)d_in[35]; const float* m2_bt = (const float*)d_in[36];
  const float* m2_g  = (const float*)d_in[37]; const float* m2_w1 = (const float*)d_in[38];
  const float* m2_w2 = (const float*)d_in[39]; const float* m2_w3 = (const float*)d_in[40];
  const float* m3_b1 = (const float*)d_in[41]; const float* m3_b2 = (const float*)d_in[42];
  const float* m3_b3 = (const float*)d_in[43]; const float* m3_bt = (const float*)d_in[44];
  const float* m3_g  = (const float*)d_in[45]; const float* m3_w1 = (const float*)d_in[46];
  const float* m3_w2 = (const float*)d_in[47]; const float* m3_w3 = (const float*)d_in[48];
  const float* paa_b  = (const float*)d_in[49]; const float* paa_bt = (const float*)d_in[50];
  const float* paa_g  = (const float*)d_in[51]; const float* paa_w  = (const float*)d_in[52];
  const float* pat_b  = (const float*)d_in[53]; const float* pat_bt = (const float*)d_in[54];
  const float* pat_g  = (const float*)d_in[55]; const float* pat_w  = (const float*)d_in[56];
  const float* root1 = (const float*)d_in[57];
  const float* root2 = (const float*)d_in[58];
  const float* root3 = (const float*)d_in[59];

  // workspace carve-up (256B aligned slabs)
  char* cur = (char*)d_ws;
  auto take = [&](size_t bytes) { char* p = cur; cur += (bytes + 255) & ~(size_t)255; return p; };
  _Float16* h1 = (_Float16*)take((size_t)E_AA2AT * 128 * 2);
  _Float16* h2 = (_Float16*)take((size_t)E_AT2AA * 128 * 2);
  _Float16* h3 = (_Float16*)take((size_t)E_AT2AT * 128 * 2);
  _Float16* w1t1 = (_Float16*)take(128 * 32 * 2);
  _Float16* w1t2 = (_Float16*)take(128 * 32 * 2);
  _Float16* w1t3 = (_Float16*)take(128 * 32 * 2);
  _Float16* w2t1 = (_Float16*)take(128 * 128 * 2);
  _Float16* w2t2 = (_Float16*)take(128 * 128 * 2);
  _Float16* w2t3 = (_Float16*)take(128 * 128 * 2);
  _Float16* w3bt1 = (_Float16*)take(128 * 1024 * 2);
  _Float16* w3bt2 = (_Float16*)take(128 * 1024 * 2);
  _Float16* w3bt3 = (_Float16*)take(128 * 1024 * 2);
  _Float16* gatwt = (_Float16*)take(512 * 128 * 2);
  _Float16* gaawt = (_Float16*)take(512 * 128 * 2);
  float* agg1 = (float*)take((size_t)N_ATOM * 32 * 4);
  float* cnt1 = (float*)take((size_t)N_ATOM * 4);
  float* agg2 = (float*)take((size_t)N_AA * 32 * 4);
  float* cnt2 = (float*)take((size_t)N_AA * 4);
  float* agg3 = (float*)take((size_t)N_ATOM * 32 * 4);
  float* cnt3 = (float*)take((size_t)N_ATOM * 4);
  float* atomF = (float*)take((size_t)N_ATOM * 128 * 4);
  float* aaF   = (float*)take((size_t)N_AA * 128 * 4);
  _Float16* xpAtom = (_Float16*)take((size_t)N_ATOM * 512 * 2);
  _Float16* xpAa   = (_Float16*)take((size_t)N_AA * 512 * 2);
  float* asA = (float*)take((size_t)N_ATOM * 4 * 4);
  float* adA = (float*)take((size_t)N_ATOM * 4 * 4);
  float* asB = (float*)take((size_t)N_AA * 4 * 4);
  float* adB = (float*)take((size_t)N_AA * 4 * 4);
  unsigned* mxA = (unsigned*)take((size_t)N_ATOM * 4 * 4);
  float* denA = (float*)take((size_t)N_ATOM * 4 * 4);
  unsigned* mxB = (unsigned*)take((size_t)N_AA * 4 * 4);
  float* denB = (float*)take((size_t)N_AA * 4 * 4);
  float* exA = (float*)take((size_t)(E_AT2AT + N_ATOM) * 4 * 4);
  float* exB = (float*)take((size_t)(E_AA2AT + N_AA) * 4 * 4);
  float* goutA = (float*)take((size_t)N_ATOM * 128 * 4);
  float* goutB = (float*)take((size_t)N_AA * 128 * 4);
  float* pooled = (float*)take(256 * 4);

  // ---- zero / init accumulators (ws is poisoned once, never re-zeroed)
  auto zf = [&](float* p, long long n) { fill_f32<<<blks(n, 256), 256, 0, stream>>>(p, 0.f, (int)n); };
  zf(agg1, (long long)N_ATOM * 32); zf(cnt1, N_ATOM);
  zf(agg2, (long long)N_AA * 32);   zf(cnt2, N_AA);
  zf(agg3, (long long)N_ATOM * 32); zf(cnt3, N_ATOM);
  zf(denA, (long long)N_ATOM * 4);  zf(denB, (long long)N_AA * 4);
  zf(goutA, (long long)N_ATOM * 128); zf(goutB, (long long)N_AA * 128);
  zf(pooled, 256);
  fill_u32<<<blks(N_ATOM * 4, 256), 256, 0, stream>>>(mxA, 0x007FFFFFu, N_ATOM * 4);  // key(-inf)
  fill_u32<<<blks(N_AA * 4, 256), 256, 0, stream>>>(mxB, 0x007FFFFFu, N_AA * 4);

  // ---- pre-transpose + f16-convert weights (K-major for B fragments)
  cvt_transpose_pad<<<blks(128 * 32, 256), 256, 0, stream>>>(m1_w1, w1t1, 32, 32, 128);
  cvt_transpose_pad<<<blks(128 * 32, 256), 256, 0, stream>>>(m2_w1, w1t2, 32, 32, 128);
  cvt_transpose_pad<<<blks(128 * 32, 256), 256, 0, stream>>>(m3_w1, w1t3, 16, 32, 128);
  cvt_transpose_pad<<<blks(128 * 128, 256), 256, 0, stream>>>(m1_w2, w2t1, 128, 128, 128);
  cvt_transpose_pad<<<blks(128 * 128, 256), 256, 0, stream>>>(m2_w2, w2t2, 128, 128, 128);
  cvt_transpose_pad<<<blks(128 * 128, 256), 256, 0, stream>>>(m3_w2, w2t3, 128, 128, 128);
  cvt_block_transpose32<<<blks(128 * 1024, 256), 256, 0, stream>>>(m1_w3, w3bt1);
  cvt_block_transpose32<<<blks(128 * 1024, 256), 256, 0, stream>>>(m2_w3, w3bt2);
  cvt_block_transpose32<<<blks(128 * 1024, 256), 256, 0, stream>>>(m3_w3, w3bt3);
  cvt_transpose_pad<<<blks(512 * 128, 256), 256, 0, stream>>>(gat_w, gatwt, 128, 128, 512);
  cvt_transpose_pad<<<blks(512 * 128, 256), 256, 0, stream>>>(gaa_w, gaawt, 128, 128, 512);

  // ---- edge MLP layers 1+2 (WMMA, LN fused, async weight staging)
  mlp_l1l2<32><<<blks(E_AA2AT, 16), 256, 0, stream>>>(ea1, w1t1, m1_b1, m1_g, m1_bt, w2t1, m1_b2, h1, E_AA2AT);
  mlp_l1l2<32><<<blks(E_AT2AA, 16), 256, 0, stream>>>(ea2, w1t2, m2_b1, m2_g, m2_bt, w2t2, m2_b2, h2, E_AT2AA);
  mlp_l1l2<16><<<blks(E_AT2AT, 16), 256, 0, stream>>>(ea3, w1t3, m3_b1, m3_g, m3_bt, w2t3, m3_b2, h3, E_AT2AT);

  // ---- fused layer-3 + NNConv message + scatter-mean accumulation (WMMA)
  l3_nnconv<<<blks(E_AA2AT, 16), 256, 0, stream>>>(h1, x_aa, ei1, ei1 + E_AA2AT, w3bt1, m1_b3, agg1, cnt1, E_AA2AT);
  l3_nnconv<<<blks(E_AT2AA, 16), 256, 0, stream>>>(h2, x_atom, ei2, ei2 + E_AT2AA, w3bt2, m2_b3, agg2, cnt2, E_AT2AA);
  l3_nnconv<<<blks(E_AT2AT, 16), 256, 0, stream>>>(h3, x_atom, ei3, ei3 + E_AT2AT, w3bt3, m3_b3, agg3, cnt3, E_AT2AT);

  // ---- segment-mean + root + hetero-mean + proj + ReLU + LN
  finalize_proj<<<N_ATOM, 128, 0, stream>>>(agg1, cnt1, agg3, cnt3, x_atom,
                                            root1, bias1, root3, bias3, 0.5f,
                                            pat_w, pat_b, pat_g, pat_bt, atomF, N_ATOM);
  finalize_proj<<<N_AA, 128, 0, stream>>>(agg2, cnt2, nullptr, nullptr, x_aa,
                                          root2, bias2, nullptr, nullptr, 1.0f,
                                          paa_w, paa_b, paa_g, paa_bt, aaF, N_AA);

  // ---- GAT: projection (WMMA + async B staging), logits, softmax, aggregation
  { dim3 g(blks(N_ATOM, 16), 4); gemm_xp<<<g, 256, 0, stream>>>(atomF, gatwt, xpAtom, N_ATOM); }
  { dim3 g(blks(N_AA, 16), 4);   gemm_xp<<<g, 256, 0, stream>>>(aaF, gaawt, xpAa, N_AA); }
  gat_att<<<blks((long long)N_ATOM * 4, 256), 256, 0, stream>>>(xpAtom, gat_as, gat_ad, asA, adA, N_ATOM);
  gat_att<<<blks((long long)N_AA * 4, 256), 256, 0, stream>>>(xpAa, gaa_as, gaa_ad, asB, adB, N_AA);

  int totA = (E_AT2AT + N_ATOM) * 4, totB = (E_AA2AT + N_AA) * 4;
  gat_edge_max<<<blks(totA, 256), 256, 0, stream>>>(ei3, ei3 + E_AT2AT, asA, adA, mxA, E_AT2AT, N_ATOM);
  gat_edge_max<<<blks(totB, 256), 256, 0, stream>>>(ei1, ei1 + E_AA2AT, asB, adB, mxB, E_AA2AT, N_AA);
  gat_edge_exp<<<blks(totA, 256), 256, 0, stream>>>(ei3, ei3 + E_AT2AT, asA, adA, mxA, exA, denA, E_AT2AT, N_ATOM);
  gat_edge_exp<<<blks(totB, 256), 256, 0, stream>>>(ei1, ei1 + E_AA2AT, asB, adB, mxB, exB, denB, E_AA2AT, N_AA);
  gat_edge_aggr<<<blks(totA, 256), 256, 0, stream>>>(ei3, ei3 + E_AT2AT, exA, denA, xpAtom, goutA, E_AT2AT, N_ATOM);
  gat_edge_aggr<<<blks(totB, 256), 256, 0, stream>>>(ei1, ei1 + E_AA2AT, exB, denB, xpAa, goutB, E_AA2AT, N_AA);

  // ---- bias + global mean pool + classifier
  bias_pool<<<128, 256, 0, stream>>>(goutA, gat_b, pooled, N_ATOM, 0);
  bias_pool<<<128, 256, 0, stream>>>(goutB, gaa_b, pooled, N_AA, 128);
  clf_kernel<<<1, 256, 0, stream>>>(pooled, c_w1, c_b1, c_w2, c_b2, c_w3, c_b3, (float*)d_out);
}
